// StreamingMixtralMoE_69011534512266
// MI455X (gfx1250) — matrix operand order repaired and checked by reading, not compiled, not run on hardware
//
#include <hip/hip_runtime.h>
#include <cstdint>
#include <cstddef>

#define NUM_EXPERTS 8
#define HIDDEN 2048
#define FFN 4096
#define NTOK 4096   // 2 * 2048 tokens

typedef __attribute__((ext_vector_type(16))) __bf16 v16bf;
typedef __attribute__((ext_vector_type(8)))  float  v8f;

union BFrag {
    v16bf v;
    __bf16 h[16];
    unsigned short u[16];
};

__device__ __forceinline__ unsigned short bfbits(float f) {
    __bf16 b = (__bf16)f;                  // native v_cvt to bf16 (RNE)
    return __builtin_bit_cast(unsigned short, b);
}

// ---------------- async global->LDS staging (CDNA5 path, guarded) ----------
#if defined(__has_builtin)
#  if __has_builtin(__builtin_amdgcn_global_load_async_to_lds_b32) && \
      __has_builtin(__builtin_amdgcn_s_wait_asynccnt)
#    define MOE_ASYNC_LDS 1
#  endif
#endif
#ifndef MOE_ASYNC_LDS
#  define MOE_ASYNC_LDS 0
#endif

__device__ __forceinline__ void stage_a32(const unsigned short* gsrc, unsigned short* ldst) {
#if MOE_ASYNC_LDS
    // memory -> LDS without VGPR round-trip; tracked by ASYNCcnt
    __builtin_amdgcn_global_load_async_to_lds_b32(
        (__attribute__((address_space(1))) int*)(uintptr_t)gsrc,
        (__attribute__((address_space(3))) int*)(uint32_t)(uintptr_t)ldst,
        0, 0);
#else
    uint32_t v32 = *(const uint32_t*)gsrc;
    ldst[0] = (unsigned short)(v32 & 0xFFFFu);
    ldst[1] = (unsigned short)(v32 >> 16);
#endif
}

__device__ __forceinline__ void wait_async_stage() {
#if MOE_ASYNC_LDS
    __builtin_amdgcn_s_wait_asynccnt(0);
#endif
}

// ---------------------------------------------------------------- zero output
__global__ void moe_zero(float* __restrict__ out, int n) {
    int i = blockIdx.x * 256 + threadIdx.x;
    if (i < n) out[i] = 0.0f;
}

// ---------------------------------------------------------------- router
__global__ void moe_router(const float* __restrict__ x,
                           const float* __restrict__ wr,
                           float* __restrict__ combine) {
    const int tid  = threadIdx.x;
    const int lane = tid & 31;
    const int wave = tid >> 5;
    const int t    = blockIdx.x * 8 + wave;

    float acc[NUM_EXPERTS];
#pragma unroll
    for (int e = 0; e < NUM_EXPERTS; ++e) acc[e] = 0.0f;

    const float* xr = x + (size_t)t * HIDDEN;
    for (int h = lane; h < HIDDEN; h += 32) {
        float xv = xr[h];
        const float* w = wr + (size_t)h * NUM_EXPERTS;
#pragma unroll
        for (int e = 0; e < NUM_EXPERTS; ++e) acc[e] += xv * w[e];
    }
#pragma unroll
    for (int e = 0; e < NUM_EXPERTS; ++e) {
#pragma unroll
        for (int off = 16; off > 0; off >>= 1)
            acc[e] += __shfl_xor(acc[e], off, 32);
    }
    if (lane == 0) {
        int i1 = 0;
#pragma unroll
        for (int e = 1; e < NUM_EXPERTS; ++e) if (acc[e] > acc[i1]) i1 = e;
        int i2 = (i1 == 0) ? 1 : 0;
#pragma unroll
        for (int e = 0; e < NUM_EXPERTS; ++e)
            if (e != i1 && acc[e] > acc[i2]) i2 = e;
        float wA = 1.0f / (1.0f + __expf(acc[i2] - acc[i1]));
        float wB = 1.0f - wA;
        float* c = combine + (size_t)t * NUM_EXPERTS;
#pragma unroll
        for (int e = 0; e < NUM_EXPERTS; ++e) c[e] = 0.0f;
        c[i1] = wA;
        c[i2] = wB;
    }
}

// ---------------------------------------------------------------- compaction
__global__ void moe_compact(const float* __restrict__ combine,
                            int* __restrict__ lists,
                            int* __restrict__ counts) {
    const int e    = blockIdx.x;
    const int tid  = threadIdx.x;
    const int lane = tid & 31;
    const int wave = tid >> 5;

    __shared__ int s_wsum[8];
    __shared__ int s_woff[8];
    __shared__ int s_base;
    if (tid == 0) s_base = 0;
    __syncthreads();

    for (int c0 = 0; c0 < NTOK; c0 += 256) {
        int t = c0 + tid;
        bool pred = combine[(size_t)t * NUM_EXPERTS + e] > 0.0f;
        unsigned long long bal = __ballot(pred);
        unsigned mask = (unsigned)bal;
        int prefix = __popc(mask & ((1u << lane) - 1u));
        if (lane == 0) s_wsum[wave] = __popc(mask);
        __syncthreads();
        if (tid == 0) {
            int run = s_base;
#pragma unroll
            for (int w = 0; w < 8; ++w) { s_woff[w] = run; run += s_wsum[w]; }
            s_base = run;
        }
        __syncthreads();
        if (pred) lists[e * NTOK + s_woff[wave] + prefix] = t;
        __syncthreads();
    }
    if (tid == 0) counts[e] = s_base;
}

// ---------------------------------------------------------------- scan
__global__ void moe_scan(const int* __restrict__ counts, int* __restrict__ bases) {
    if (threadIdx.x == 0) {
        int run = 0;
#pragma unroll
        for (int e = 0; e < NUM_EXPERTS; ++e) { bases[e] = run; run += counts[e]; }
    }
}

// ---------------------------------------------------------------- x -> bf16
__global__ void moe_cvt_x(const float* __restrict__ x, unsigned short* __restrict__ xb) {
    size_t i = ((size_t)blockIdx.x * 256 + threadIdx.x) * 4;
    float4 t = *(const float4*)(x + i);
    ushort4 o;
    o.x = bfbits(t.x); o.y = bfbits(t.y); o.z = bfbits(t.z); o.w = bfbits(t.w);
    *(ushort4*)(xb + i) = o;
}

// ---------------------------------------------------------------- fragment helpers
__device__ __forceinline__ void load_bfrag(BFrag& b, const float* __restrict__ p) {
    const float4* q = (const float4*)p;
#pragma unroll
    for (int i = 0; i < 4; ++i) {
        float4 t = q[i];
        b.h[i * 4 + 0] = (__bf16)t.x;
        b.h[i * 4 + 1] = (__bf16)t.y;
        b.h[i * 4 + 2] = (__bf16)t.z;
        b.h[i * 4 + 3] = (__bf16)t.w;
    }
}

__device__ __forceinline__ void load_afrag(BFrag& a, const unsigned short sa[16][34], int lane) {
    const int m  = lane & 15;
    const int h8 = (lane >> 4) * 8;
#pragma unroll
    for (int j = 0; j < 16; ++j) {
        int k = j + 2 * h8;
        a.u[j] = sa[m][k];
    }
}

#define WMMA_BF16(A, B, C) \
    __builtin_amdgcn_wmma_f32_16x16x32_bf16(false, (A).v, false, (B).v, (short)0, (C), false, false)

// ---------------------------------------------------------------- FFN stage 1
// h = silu(x W1^T) * (x W3^T); WG = 256 thr (8 waves)
// tile = 16 rows x 256 F cols; 2 N-tiles/wave; k-loop unrolled x2 (8 WMMA/iter)
__global__ void __launch_bounds__(256)
moe_ffn1(const unsigned short* __restrict__ xb,
         const float* __restrict__ w1,
         const float* __restrict__ w3,
         const int* __restrict__ lists,
         const int* __restrict__ counts,
         const int* __restrict__ bases,
         unsigned short* __restrict__ hbuf) {
    const int e   = blockIdx.z;
    const int cnt = counts[e];
    const int rowtile = blockIdx.y;
    if (rowtile * 16 >= cnt) return;

    const int fbase = blockIdx.x * 256;
    const int tid  = threadIdx.x;
    const int lane = tid & 31;
    const int wave = tid >> 5;
    const int n    = lane & 15;
    const int half = lane >> 4;

    __shared__ int s_tok[16];
    __shared__ unsigned short s_a[2][16][34];   // double-buffered 16x32 bf16 A tile

    if (tid < 16) {
        int r = rowtile * 16 + tid;
        if (r >= cnt) r = cnt - 1;
        s_tok[tid] = lists[e * NTOK + r];
    }
    __syncthreads();

    const int f0 = fbase + wave * 32 + n;               // first N-tile column
    const float* w1pa = w1 + ((size_t)e * FFN + f0) * HIDDEN + half * 16;
    const float* w1pb = w1pa + (size_t)16 * HIDDEN;
    const float* w3pa = w3 + ((size_t)e * FFN + f0) * HIDDEN + half * 16;
    const float* w3pb = w3pa + (size_t)16 * HIDDEN;

    v8f acc1a = {}, acc1b = {}, acc3a = {}, acc3b = {};

    const int arow = tid >> 4;   // 0..15
    const int akp  = tid & 15;   // k-pair index
    const size_t abase = (size_t)s_tok[arow] * HIDDEN + akp * 2;
    unsigned short* l0 = &s_a[0][arow][akp * 2];
    unsigned short* l1 = &s_a[1][arow][akp * 2];

    stage_a32(xb + abase, l0);   // prologue: k=0 into buffer 0

#define FFN1_STEP(BUF, KOFF)                                        \
    {                                                               \
        BFrag a;                                                    \
        load_afrag(a, s_a[BUF], lane);                              \
        BFrag b1a, b1b, b3a, b3b;                                   \
        load_bfrag(b1a, w1pa + (KOFF));                             \
        load_bfrag(b1b, w1pb + (KOFF));                             \
        load_bfrag(b3a, w3pa + (KOFF));                             \
        load_bfrag(b3b, w3pb + (KOFF));                             \
        acc1a = WMMA_BF16(a, b1a, acc1a);                           \
        acc1b = WMMA_BF16(a, b1b, acc1b);                           \
        acc3a = WMMA_BF16(a, b3a, acc3a);                           \
        acc3b = WMMA_BF16(a, b3b, acc3b);                           \
    }

    for (int k0 = 0; k0 < HIDDEN; k0 += 64) {
        wait_async_stage();
        __syncthreads();
        // k0 <= HIDDEN-64 so k0+32 is always in range
        stage_a32(xb + abase + k0 + 32, l1);
        __builtin_prefetch(w1pa + k0 + 32, 0, 0);
        __builtin_prefetch(w1pb + k0 + 32, 0, 0);
        __builtin_prefetch(w3pa + k0 + 32, 0, 0);
        __builtin_prefetch(w3pb + k0 + 32, 0, 0);
        FFN1_STEP(0, k0);

        wait_async_stage();
        __syncthreads();
        if (k0 + 64 < HIDDEN) {
            stage_a32(xb + abase + k0 + 64, l0);
            __builtin_prefetch(w1pa + k0 + 64, 0, 0);
            __builtin_prefetch(w1pb + k0 + 64, 0, 0);
            __builtin_prefetch(w3pa + k0 + 64, 0, 0);
            __builtin_prefetch(w3pb + k0 + 64, 0, 0);
        }
        FFN1_STEP(1, k0 + 32);
    }
#undef FFN1_STEP

    const int rowglob0 = bases[e] + rowtile * 16;
#pragma unroll
    for (int v = 0; v < 8; ++v) {
        int m = v + half * 8;
        int r = rowtile * 16 + m;
        if (r < cnt) {
            size_t rb = (size_t)(rowglob0 + m) * FFN;
            float g0 = acc1a[v];
            float h0 = g0 * __builtin_amdgcn_rcpf(1.0f + __expf(-g0)) * acc3a[v];
            hbuf[rb + f0] = bfbits(h0);
            float g1 = acc1b[v];
            float h1 = g1 * __builtin_amdgcn_rcpf(1.0f + __expf(-g1)) * acc3b[v];
            hbuf[rb + f0 + 16] = bfbits(h1);
        }
    }
}

// ---------------------------------------------------------------- FFN stage 2
// out[token] += combine * (h W2^T); tile = 16 rows x 256 H cols; unrolled x2
__global__ void __launch_bounds__(256)
moe_ffn2(const unsigned short* __restrict__ hbuf,
         const float* __restrict__ w2,
         const int* __restrict__ lists,
         const int* __restrict__ counts,
         const int* __restrict__ bases,
         const float* __restrict__ combine,
         float* __restrict__ out) {
    const int e   = blockIdx.z;
    const int cnt = counts[e];
    const int rowtile = blockIdx.y;
    if (rowtile * 16 >= cnt) return;

    const int hbase = blockIdx.x * 256;
    const int tid  = threadIdx.x;
    const int lane = tid & 31;
    const int wave = tid >> 5;
    const int n    = lane & 15;
    const int half = lane >> 4;

    __shared__ int s_tok[16];
    __shared__ unsigned short s_a[2][16][34];

    if (tid < 16) {
        int r = rowtile * 16 + tid;
        if (r >= cnt) r = cnt - 1;
        s_tok[tid] = lists[e * NTOK + r];
    }
    __syncthreads();

    const int hc0 = hbase + wave * 32 + n;
    const float* w2pa = w2 + ((size_t)e * HIDDEN + hc0) * FFN + half * 16;
    const float* w2pb = w2pa + (size_t)16 * FFN;

    v8f acca = {}, accb = {};

    const int arow = tid >> 4;
    const int akp  = tid & 15;
    int arc = rowtile * 16 + arow;
    if (arc >= cnt) arc = cnt - 1;                    // clamp tail rows
    const size_t abase = (size_t)(bases[e] + arc) * FFN + akp * 2;
    unsigned short* l0 = &s_a[0][arow][akp * 2];
    unsigned short* l1 = &s_a[1][arow][akp * 2];

    stage_a32(hbuf + abase, l0);

#define FFN2_STEP(BUF, KOFF)                                        \
    {                                                               \
        BFrag a;                                                    \
        load_afrag(a, s_a[BUF], lane);                              \
        BFrag ba, bb;                                               \
        load_bfrag(ba, w2pa + (KOFF));                              \
        load_bfrag(bb, w2pb + (KOFF));                              \
        acca = WMMA_BF16(a, ba, acca);                              \
        accb = WMMA_BF16(a, bb, accb);                              \
    }

    for (int k0 = 0; k0 < FFN; k0 += 64) {
        wait_async_stage();
        __syncthreads();
        stage_a32(hbuf + abase + k0 + 32, l1);
        __builtin_prefetch(w2pa + k0 + 32, 0, 0);
        __builtin_prefetch(w2pb + k0 + 32, 0, 0);
        FFN2_STEP(0, k0);

        wait_async_stage();
        __syncthreads();
        if (k0 + 64 < FFN) {
            stage_a32(hbuf + abase + k0 + 64, l0);
            __builtin_prefetch(w2pa + k0 + 64, 0, 0);
            __builtin_prefetch(w2pb + k0 + 64, 0, 0);
        }
        FFN2_STEP(1, k0 + 32);
    }
#undef FFN2_STEP

#pragma unroll
    for (int v = 0; v < 8; ++v) {
        int m = v + half * 8;
        int r = rowtile * 16 + m;
        if (r < cnt) {
            int token = s_tok[m];
            float cw = combine[(size_t)token * NUM_EXPERTS + e];
            float* orow = out + (size_t)token * HIDDEN;
            atomicAdd(&orow[hc0],      acca[v] * cw);
            atomicAdd(&orow[hc0 + 16], accb[v] * cw);
        }
    }
}

// ---------------------------------------------------------------- launch
extern "C" void kernel_launch(void* const* d_in, const int* in_sizes, int n_in,
                              void* d_out, int out_size, void* d_ws, size_t ws_size,
                              hipStream_t stream) {
    (void)in_sizes; (void)n_in; (void)ws_size;
    const float* x  = (const float*)d_in[0];
    const float* wr = (const float*)d_in[1];
    const float* w1 = (const float*)d_in[2];
    const float* w2 = (const float*)d_in[3];
    const float* w3 = (const float*)d_in[4];
    float* out = (float*)d_out;

    char* ws = (char*)d_ws;
    float* combine = (float*)ws;                                   // 128 KB
    int*   lists   = (int*)(ws + (size_t)(128 << 10));             // 128 KB
    int*   counts  = (int*)(ws + (size_t)(256 << 10));             // 32 B
    int*   bases   = (int*)(ws + (size_t)(256 << 10) + 64);        // 32 B
    unsigned short* xb   = (unsigned short*)(ws + (size_t)(257 << 10));           // 16 MB
    unsigned short* hbuf = (unsigned short*)(ws + (size_t)(257 << 10)
                                             + (size_t)NTOK * HIDDEN * 2);        // 64 MB

    moe_zero<<<(out_size + 255) / 256, 256, 0, stream>>>(out, out_size);
    moe_router<<<NTOK / 8, 256, 0, stream>>>(x, wr, combine);
    moe_compact<<<NUM_EXPERTS, 256, 0, stream>>>(combine, lists, counts);
    moe_scan<<<1, 32, 0, stream>>>(counts, bases);
    moe_cvt_x<<<(NTOK * HIDDEN) / (256 * 4), 256, 0, stream>>>(x, xb);

    dim3 g1(FFN / 256, NTOK / 16, NUM_EXPERTS);
    moe_ffn1<<<g1, 256, 0, stream>>>(xb, w1, w3, lists, counts, bases, hbuf);

    dim3 g2(HIDDEN / 256, NTOK / 16, NUM_EXPERTS);
    moe_ffn2<<<g2, 256, 0, stream>>>(hbuf, w2, lists, counts, bases, combine, out);
}
